// MessageModel_2267742732913
// MI455X (gfx1250) — matrix operand order repaired and checked by reading, not compiled, not run on hardware
//
#include <hip/hip_runtime.h>

#define E_EDGES 320000
#define N_NODES 10000
#define D_NODE  256
#define D_EDGE  256
#define D_IN    512
#define D_HID   512
#define D_OUT   256

typedef __attribute__((ext_vector_type(16))) __bf16 v16bf;
typedef __attribute__((ext_vector_type(8)))  float  v8f;

#if defined(__has_builtin)
#if __has_builtin(__builtin_amdgcn_s_wait_asynccnt)
#define WAIT_ASYNC0() __builtin_amdgcn_s_wait_asynccnt(0)
#endif
#endif
#ifndef WAIT_ASYNC0
#define WAIT_ASYNC0() asm volatile("s_wait_asynccnt 0" ::: "memory")
#endif

__device__ __forceinline__ unsigned short f2bf(float f) {
  unsigned u = __float_as_uint(f);
  u += 0x7FFFu + ((u >> 16) & 1u);   // round-to-nearest-even
  return (unsigned short)(u >> 16);
}

// async copy of 16 bytes per lane: global -> LDS, tracked by ASYNCcnt
__device__ __forceinline__ void async_cp16(unsigned lds_byte_addr, const void* gaddr) {
  asm volatile("global_load_async_to_lds_b128 %0, %1, off"
               :: "v"(lds_byte_addr), "v"(gaddr) : "memory");
}

__global__ void zero_out_kernel(float* __restrict__ out, int n) {
  int i = blockIdx.x * blockDim.x + threadIdx.x;
  if (i < n) out[i] = 0.0f;
}

// W: [rowsK][colsN] row-major f32  ->  WT: [colsN][rowsK] bf16
__global__ void transpose_bf16_kernel(const float* __restrict__ W,
                                      unsigned short* __restrict__ WT,
                                      int rowsK, int colsN) {
  int i = blockIdx.x * blockDim.x + threadIdx.x;
  if (i >= rowsK * colsN) return;
  int n = i / rowsK;
  int k = i - n * rowsK;
  WT[n * rowsK + k] = f2bf(W[k * colsN + n]);
}

extern "C" __global__ void __launch_bounds__(128)
msg_mlp_kernel(const float* __restrict__ x_in,
               const int*   __restrict__ eidx,   // [2][E]: rows then cols
               const float* __restrict__ eattr,
               const unsigned short* __restrict__ W1T, // [512 n][512 k] bf16
               const float* __restrict__ b1,
               const unsigned short* __restrict__ W2T, // [256 n][512 k] bf16
               const float* __restrict__ b2,
               float* __restrict__ out) {
  extern __shared__ unsigned short smem[];
  const int AST = 520;                         // padded row stride (ushorts)
  unsigned short* ldsA  = smem;                // 64 x 520
  unsigned short* ldsW1 = ldsA  + 64 * AST;    // 32 x 520
  unsigned short* ldsW2 = ldsW1 + 32 * AST;    // 256 x 40
  unsigned short* ldsH  = ldsW2 + 256 * 40;    // 4 waves x 16 x 40
  float* b1s = (float*)(ldsH + 4 * 16 * 40);   // 512
  float* b2s = b1s + 512;                      // 256

  const int tid  = threadIdx.x;
  const int lane = tid & 31;
  const int wave = tid >> 5;
  const int hl   = (lane & 16) ? 1 : 0;
  const int l15  = lane & 15;
  const int blkEdge = blockIdx.x * 64;

  const unsigned ldsW1base = (unsigned)(uintptr_t)ldsW1;
  const unsigned ldsW2base = (unsigned)(uintptr_t)ldsW2;

  for (int i = tid; i < 512; i += 128) b1s[i] = b1[i];
  for (int i = tid; i < 256; i += 128) b2s[i] = b2[i];

  // ---- stage A tile: 64 edges x (x_in[col] || edge_attr) as bf16 ----
  {
    int le   = tid >> 1;
    int half = tid & 1;
    int e    = blkEdge + le;
    const float4* src;
    unsigned short* dst = ldsA + le * AST + half * 256;
    if (half == 0) {
      int col = eidx[E_EDGES + e];
      src = (const float4*)(x_in + (long)col * D_NODE);
    } else {
      src = (const float4*)(eattr + (long)e * D_EDGE);
    }
    #pragma unroll 4
    for (int j = 0; j < 64; ++j) {
      float4 f = src[j];
      ushort4 h;
      h.x = f2bf(f.x); h.y = f2bf(f.y); h.z = f2bf(f.z); h.w = f2bf(f.w);
      *(ushort4*)(dst + j * 4) = h;
    }
  }
  __syncthreads();

  v8f acc[16];
  v8f zero8 = {0.f,0.f,0.f,0.f,0.f,0.f,0.f,0.f};
  #pragma unroll
  for (int t = 0; t < 16; ++t) acc[t] = zero8;

  unsigned short* myH = ldsH + wave * 16 * 40;
  const unsigned short* aRow = ldsA + (wave * 16 + l15) * AST;

  for (int c = 0; c < 16; ++c) {
    // ---- async DMA: W1 chunk [32 n][512 k] (32 KB), W2 chunk [256 n][32 k] (16 KB) ----
    {
      const char* w1src = (const char*)(W1T + (size_t)c * 32 * 512);
      #pragma unroll
      for (int j = 0; j < 16; ++j) {
        int u   = (wave * 16 + j) * 32 + lane;   // 16-byte units, 0..2047
        int row = u >> 6;                        // 64 units per 1024B source row
        int col = u & 63;
        unsigned lds = ldsW1base + (unsigned)(row * 1040 + col * 16);
        async_cp16(lds, w1src + (size_t)u * 16);
      }
      #pragma unroll
      for (int j = 0; j < 8; ++j) {
        int u   = (wave * 8 + j) * 32 + lane;    // 16-byte units, 0..1023
        int row = u >> 2;                        // 4 units per 64B slice
        int col = u & 3;
        unsigned lds = ldsW2base + (unsigned)(row * 80 + col * 16);
        const char* g = (const char*)(W2T + (size_t)row * 512 + c * 32) + col * 16;
        async_cp16(lds, g);
      }
      WAIT_ASYNC0();
    }
    __syncthreads();

    // ---- GEMM1: H[16x32] = A[16x512] * W1chunk[512x32] ----
    v8f h0 = zero8, h1 = zero8;
    for (int kk = 0; kk < 16; ++kk) {
      union { v16bf v; uint4 q[2]; } a, bA, bB;
      int ka = kk * 32 + (hl ? 8 : 0);
      a.q[0] = *(const uint4*)(aRow + ka);
      a.q[1] = *(const uint4*)(aRow + ka + 16);
      int kb = kk * 32 + (hl ? 8 : 0);
      const unsigned short* w1r0 = ldsW1 + l15 * AST + kb;
      const unsigned short* w1r1 = ldsW1 + (16 + l15) * AST + kb;
      bA.q[0] = *(const uint4*)(w1r0);
      bA.q[1] = *(const uint4*)(w1r0 + 16);
      bB.q[0] = *(const uint4*)(w1r1);
      bB.q[1] = *(const uint4*)(w1r1 + 16);
      h0 = __builtin_amdgcn_wmma_f32_16x16x32_bf16(false, a.v, false, bA.v,
                                                   (short)0, h0, false, false);
      h1 = __builtin_amdgcn_wmma_f32_16x16x32_bf16(false, a.v, false, bB.v,
                                                   (short)0, h1, false, false);
    }

    // ---- bias + ReLU, C-layout -> A-layout via per-wave LDS buffer ----
    {
      int hr = hl ? 8 : 0;
      float bb0 = b1s[c * 32 + l15];
      float bb1 = b1s[c * 32 + 16 + l15];
      #pragma unroll
      for (int v = 0; v < 8; ++v) {
        float f0 = h0[v] + bb0; f0 = f0 > 0.f ? f0 : 0.f;
        float f1 = h1[v] + bb1; f1 = f1 > 0.f ? f1 : 0.f;
        myH[(v + hr) * 40 + l15]      = f2bf(f0);
        myH[(v + hr) * 40 + 16 + l15] = f2bf(f1);
      }
    }
    union { v16bf v; uint4 q[2]; } a2;
    {
      const unsigned short* hrow = myH + l15 * 40 + (hl ? 8 : 0);
      a2.q[0] = *(const uint4*)(hrow);
      a2.q[1] = *(const uint4*)(hrow + 16);
    }

    // ---- GEMM2: out[16x256] += H[16x32] * W2chunk[32x256] ----
    #pragma unroll
    for (int t = 0; t < 16; ++t) {
      union { v16bf v; uint4 q[2]; } b;
      const unsigned short* w2r = ldsW2 + (t * 16 + l15) * 40 + (hl ? 8 : 0);
      b.q[0] = *(const uint4*)(w2r);
      b.q[1] = *(const uint4*)(w2r + 16);
      acc[t] = __builtin_amdgcn_wmma_f32_16x16x32_bf16(false, a2.v, false, b.v,
                                                       (short)0, acc[t], false, false);
    }
    __syncthreads();
  }

  // ---- bias2 + scatter-add to destination nodes ----
  #pragma unroll
  for (int v = 0; v < 8; ++v) {
    int e = blkEdge + wave * 16 + v + (hl ? 8 : 0);
    int r = eidx[e];
    float* dst = out + (long)r * D_OUT + l15;
    #pragma unroll
    for (int t = 0; t < 16; ++t) {
      float val = acc[t][v] + b2s[t * 16 + l15];
      unsafeAtomicAdd(dst + t * 16, val);
    }
  }
}

extern "C" void kernel_launch(void* const* d_in, const int* in_sizes, int n_in,
                              void* d_out, int out_size, void* d_ws, size_t ws_size,
                              hipStream_t stream) {
  (void)in_sizes; (void)n_in; (void)out_size; (void)ws_size;
  const float* x_in  = (const float*)d_in[1];
  const int*   eidx  = (const int*)d_in[2];
  const float* eattr = (const float*)d_in[3];
  const float* W1    = (const float*)d_in[4];
  const float* b1    = (const float*)d_in[5];
  const float* W2    = (const float*)d_in[6];
  const float* b2    = (const float*)d_in[7];
  float* out = (float*)d_out;

  unsigned short* W1T = (unsigned short*)d_ws;       // 512*512 bf16
  unsigned short* W2T = W1T + 512 * 512;             // 256*512 bf16

  zero_out_kernel<<<(N_NODES * D_OUT + 255) / 256, 256, 0, stream>>>(out, N_NODES * D_OUT);
  transpose_bf16_kernel<<<(512 * 512 + 255) / 256, 256, 0, stream>>>(W1, W1T, 512, 512);
  transpose_bf16_kernel<<<(512 * 256 + 255) / 256, 256, 0, stream>>>(W2, W2T, 512, 256);

  size_t smem = (size_t)(64 * 520 + 32 * 520 + 256 * 40 + 4 * 16 * 40) * sizeof(unsigned short)
              + (size_t)(512 + 256) * sizeof(float);   // 128,512 B
  msg_mlp_kernel<<<E_EDGES / 64, 128, smem, stream>>>(x_in, eidx, eattr,
                                                      W1T, b1, W2T, b2, out);
}
